// SetAbstraction_12214886990744
// MI455X (gfx1250) — matrix operand order repaired and checked by reading, not compiled, not run on hardware
//
#include <hip/hip_runtime.h>
#include <hip/hip_bf16.h>
#include <stdint.h>

typedef __bf16 bf16_t;
typedef __bf16 v16bf __attribute__((ext_vector_type(16)));
typedef float  v8f   __attribute__((ext_vector_type(8)));

#define BB 16
#define NN 2048
#define KK 32
#define SS 512
#define PP (BB*SS*KK)   /* 262144 grouped positions */
#define PC (BB*SS)      /* 8192 center positions   */
#define CH 288          /* attention hidden 264 padded to 288 (18 n-tiles) */

// ---------------------------------------------------------------------------
// Farthest point sampling: one block per batch, sequential 512 iterations.
// ---------------------------------------------------------------------------
__global__ void k_fps(const float* __restrict__ xyz, int* __restrict__ fps_idx) {
  __shared__ float dist[NN];
  __shared__ float rv[256];
  __shared__ int   ri[256];
  __shared__ int   s_far;
  const int b = blockIdx.x;
  const int t = threadIdx.x;
  const float* px = xyz + (size_t)b * NN * 3;
  for (int j = t; j < NN; j += 256) dist[j] = 1e10f;
  if (t == 0) s_far = 0;
  __syncthreads();
  for (int it = 0; it < SS; ++it) {
    int far = s_far;
    if (t == 0) fps_idx[b * SS + it] = far;
    float cx = px[far*3+0], cy = px[far*3+1], cz = px[far*3+2];
    float bm = -1.0f; int bi = 0;
    for (int j = t; j < NN; j += 256) {
      float dx = px[j*3+0]-cx, dy = px[j*3+1]-cy, dz = px[j*3+2]-cz;
      float d = dx*dx + dy*dy + dz*dz;
      float old = dist[j];
      d = (d < old) ? d : old;
      dist[j] = d;
      if (d > bm || (d == bm && j < bi)) { bm = d; bi = j; }
    }
    rv[t] = bm; ri[t] = bi;
    __syncthreads();
    for (int off = 128; off > 0; off >>= 1) {
      if (t < off) {
        float v2 = rv[t+off]; int i2 = ri[t+off];
        if (v2 > rv[t] || (v2 == rv[t] && i2 < ri[t])) { rv[t] = v2; ri[t] = i2; }
      }
      __syncthreads();
    }
    if (t == 0) s_far = ri[0];
    __syncthreads();
  }
}

// ---------------------------------------------------------------------------
// KNN (top-32 nearest, self excluded) only for the S sampled points.
// ---------------------------------------------------------------------------
__global__ void k_knn(const float* __restrict__ xyz, const int* __restrict__ fps_idx,
                      int* __restrict__ knn) {
  int q = blockIdx.x * blockDim.x + threadIdx.x;
  if (q >= PC) return;
  int b = q / SS;
  const float* px = xyz + (size_t)b * NN * 3;
  int ci = fps_idx[q];
  float qx = px[ci*3+0], qy = px[ci*3+1], qz = px[ci*3+2];
  float dl[KK]; int il[KK];
  for (int t = 0; t < KK; ++t) { dl[t] = 3.4e38f; il[t] = 0; }
  for (int j = 0; j < NN; ++j) {
    if (j == ci) continue;
    float dx = px[j*3+0]-qx, dy = px[j*3+1]-qy, dz = px[j*3+2]-qz;
    float d2 = dx*dx + dy*dy + dz*dz;
    if (d2 < dl[KK-1]) {
      int t = KK - 1;
      while (t > 0 && dl[t-1] > d2) { dl[t] = dl[t-1]; il[t] = il[t-1]; --t; }
      dl[t] = d2; il[t] = j;
    }
  }
  int* out = knn + (size_t)q * KK;
  for (int t = 0; t < KK; ++t) out[t] = il[t];
}

// ---------------------------------------------------------------------------
// Weight packing: f32 [Co,Ci] -> bf16 [Cop,Cip] zero padded. Vec padding.
// ---------------------------------------------------------------------------
__global__ void k_pack_w(bf16_t* __restrict__ dst, const float* __restrict__ src,
                         int Co, int Ci, int Cop, int Cip) {
  size_t i = (size_t)blockIdx.x * 256 + threadIdx.x;
  size_t total = (size_t)Cop * Cip;
  if (i >= total) return;
  int o = (int)(i / Cip), c = (int)(i % Cip);
  float v = (o < Co && c < Ci) ? src[(size_t)o * Ci + c] : 0.0f;
  dst[i] = (bf16_t)v;
}

__global__ void k_pad_vec(float* __restrict__ dst, const float* __restrict__ src,
                          int Co, int Cop) {
  int i = blockIdx.x * 256 + threadIdx.x;
  if (i >= Cop) return;
  dst[i] = (i < Co) ? src[i] : 0.0f;
}

// ---------------------------------------------------------------------------
// Build grouped input X0 [PP, 96] (82 real + 14 zero pad) in bf16.
// concat order: g_xyz_rel, g_eula_rel, g_near, cn, g_meta, cm, g_fea.
// ---------------------------------------------------------------------------
__global__ void k_build_x0(const float* __restrict__ xyz, const float* __restrict__ eula,
                           const float* __restrict__ nearb, const float* __restrict__ meta,
                           const float* __restrict__ fea, const int* __restrict__ fps_idx,
                           const int* __restrict__ knn, bf16_t* __restrict__ X0) {
  int p = blockIdx.x * 256 + threadIdx.x;
  if (p >= PP) return;
  int bs = p / KK, k = p % KK;
  int b = bs / SS;
  int ci = fps_idx[bs];
  int j  = knn[(size_t)bs * KK + k];
  size_t cb = (size_t)b * NN + ci;
  size_t jb = (size_t)b * NN + j;
  bf16_t* r = X0 + (size_t)p * 96;
  for (int d = 0; d < 3; ++d)  r[d]      = (bf16_t)(xyz[jb*3+d]  - xyz[cb*3+d]);
  for (int d = 0; d < 3; ++d)  r[3+d]    = (bf16_t)(eula[jb*3+d] - eula[cb*3+d]);
  for (int d = 0; d < 2; ++d)  r[6+d]    = (bf16_t)nearb[jb*2+d];
  for (int d = 0; d < 2; ++d)  r[8+d]    = (bf16_t)nearb[cb*2+d];
  for (int d = 0; d < 4; ++d)  r[10+d]   = (bf16_t)meta[jb*4+d];
  for (int d = 0; d < 4; ++d)  r[14+d]   = (bf16_t)meta[cb*4+d];
  for (int d = 0; d < 64; ++d) r[18+d]   = (bf16_t)fea[jb*64+d];
  for (int d = 82; d < 96; ++d) r[d]     = (bf16_t)0.0f;
}

// Center feature input Xc0 [PC, 96]: zeros(6), cn, cn, cm, cm, c_fea, pad.
__global__ void k_build_xc0(const float* __restrict__ nearb, const float* __restrict__ meta,
                            const float* __restrict__ fea, const int* __restrict__ fps_idx,
                            bf16_t* __restrict__ Xc0) {
  int bs = blockIdx.x * 256 + threadIdx.x;
  if (bs >= PC) return;
  int b = bs / SS;
  int ci = fps_idx[bs];
  size_t cb = (size_t)b * NN + ci;
  bf16_t* r = Xc0 + (size_t)bs * 96;
  for (int d = 0; d < 6; ++d)  r[d]    = (bf16_t)0.0f;
  for (int d = 0; d < 2; ++d)  r[6+d]  = (bf16_t)nearb[cb*2+d];
  for (int d = 0; d < 2; ++d)  r[8+d]  = (bf16_t)nearb[cb*2+d];
  for (int d = 0; d < 4; ++d)  r[10+d] = (bf16_t)meta[cb*4+d];
  for (int d = 0; d < 4; ++d)  r[14+d] = (bf16_t)meta[cb*4+d];
  for (int d = 0; d < 64; ++d) r[18+d] = (bf16_t)fea[cb*64+d];
  for (int d = 82; d < 96; ++d) r[d]   = (bf16_t)0.0f;
}

// Gathered pass-through outputs: center xyz/eula/near/meta and center_fea.
__global__ void k_outputs(const float* __restrict__ xyz, const float* __restrict__ eula,
                          const float* __restrict__ nearb, const float* __restrict__ meta,
                          const float* __restrict__ fea, const int* __restrict__ fps_idx,
                          float* __restrict__ out) {
  int bs = blockIdx.x * 256 + threadIdx.x;
  if (bs >= PC) return;
  int b = bs / SS;
  int ci = fps_idx[bs];
  size_t cb = (size_t)b * NN + ci;
  float* o1 = out;                 // [B,S,3]
  float* o2 = out + 24576;         // [B,S,3]
  float* o3 = out + 49152;         // [B,S,2]
  float* o4 = out + 65536;         // [B,S,4]
  float* o5 = out + 98304;         // [B,S,320]
  for (int d = 0; d < 3; ++d)  o1[(size_t)bs*3 + d] = xyz[cb*3+d];
  for (int d = 0; d < 3; ++d)  o2[(size_t)bs*3 + d] = eula[cb*3+d];
  for (int d = 0; d < 2; ++d)  o3[(size_t)bs*2 + d] = nearb[cb*2+d];
  for (int d = 0; d < 4; ++d)  o4[(size_t)bs*4 + d] = meta[cb*4+d];
  for (int d = 0; d < 64; ++d) o5[(size_t)bs*320 + d] = fea[cb*64+d];
}

// ---------------------------------------------------------------------------
// N-blocked WMMA GEMM: Y[P,Co] = X[P,Ci] * W[Co,Ci]^T + bias.
// One wave computes a 16(pos) x (NB*16)(cout) tile: the A fragment is loaded
// once per k-step and reused for NB v_wmma_f32_16x16x32_bf16 issues, cutting
// activation re-reads by NB.  Ci % 32 == 0, (Co/16) % NB == 0, P % 16 == 0.
// ---------------------------------------------------------------------------
union Frag { v16bf v; uint4 q[2]; };

template <int NB>
__global__ __launch_bounds__(256)
void k_gemm(const bf16_t* __restrict__ X, const bf16_t* __restrict__ W,
            const float* __restrict__ bias, bf16_t* __restrict__ Y,
            int P, int Ci, int Co) {
  const int ngroups = (Co >> 4) / NB;
  const int wid = (blockIdx.x * 256 + threadIdx.x) >> 5;
  const int total = (P >> 4) * ngroups;
  if (wid >= total) return;
  const int lane = threadIdx.x & 31;
  const int row  = lane & 15;      // A: M row ; B/D: N column
  const int hi   = lane >> 4;
  const int pt = wid / ngroups, ng = wid % ngroups;
  const int pbase = pt << 4, nbase = ng * NB * 16;
  const bf16_t* xr = X + (size_t)(pbase + row) * Ci;
  const bf16_t* wr[NB];
#pragma unroll
  for (int n = 0; n < NB; ++n)
    wr[n] = W + (size_t)(nbase + n * 16 + row) * Ci + hi * 8;
  v8f acc[NB];
#pragma unroll
  for (int n = 0; n < NB; ++n) acc[n] = (v8f){0.f,0.f,0.f,0.f,0.f,0.f,0.f,0.f};
  for (int k0 = 0; k0 < Ci; k0 += 32) {
    Frag a;
    a.q[0] = *(const uint4*)(xr + k0 + hi * 8);        // K = hi*8 .. +8
    a.q[1] = *(const uint4*)(xr + k0 + 16 + hi * 8);   // K = 16+hi*8 .. +8
#pragma unroll
    for (int n = 0; n < NB; ++n) {
      Frag bf;
      bf.q[0] = *(const uint4*)(wr[n] + k0);           // K = hi*8 .. +8
      bf.q[1] = *(const uint4*)(wr[n] + k0 + 16);      // K = 16+hi*8 .. +8
      acc[n] = __builtin_amdgcn_wmma_f32_16x16x32_bf16(false, a.v, false, bf.v,
                                                       (short)0, acc[n], false, false);
    }
  }
#pragma unroll
  for (int n = 0; n < NB; ++n) {
    const float bv = bias[nbase + n * 16 + row];
    bf16_t* yb = Y + (size_t)(pbase + hi * 8) * Co + nbase + n * 16 + row;
    for (int r = 0; r < 8; ++r)
      yb[(size_t)r * Co] = (bf16_t)(acc[n][r] + bv);
  }
}

// Same GEMM but A = (fai[bs] - psi[p]) computed on the fly (gamma head input).
template <int NB>
__global__ __launch_bounds__(256)
void k_gemm_sub(const bf16_t* __restrict__ Xp, const bf16_t* __restrict__ F,
                const bf16_t* __restrict__ W, const float* __restrict__ bias,
                bf16_t* __restrict__ Y, int P, int Ci, int Co) {
  const int ngroups = (Co >> 4) / NB;
  const int wid = (blockIdx.x * 256 + threadIdx.x) >> 5;
  const int total = (P >> 4) * ngroups;
  if (wid >= total) return;
  const int lane = threadIdx.x & 31;
  const int row  = lane & 15;
  const int hi   = lane >> 4;
  const int pt = wid / ngroups, ng = wid % ngroups;
  const int pbase = pt << 4, nbase = ng * NB * 16;
  const int prow = pbase + row;
  const bf16_t* xr = Xp + (size_t)prow * Ci;
  const bf16_t* fr = F  + (size_t)(prow >> 5) * Ci;   // bs = p / K (K==32)
  const bf16_t* wr[NB];
#pragma unroll
  for (int n = 0; n < NB; ++n)
    wr[n] = W + (size_t)(nbase + n * 16 + row) * Ci + hi * 8;
  v8f acc[NB];
#pragma unroll
  for (int n = 0; n < NB; ++n) acc[n] = (v8f){0.f,0.f,0.f,0.f,0.f,0.f,0.f,0.f};
  for (int k0 = 0; k0 < Ci; k0 += 32) {
    Frag a, xa, fa;
    xa.q[0] = *(const uint4*)(xr + k0 + hi * 8);
    xa.q[1] = *(const uint4*)(xr + k0 + 16 + hi * 8);
    fa.q[0] = *(const uint4*)(fr + k0 + hi * 8);
    fa.q[1] = *(const uint4*)(fr + k0 + 16 + hi * 8);
#pragma unroll
    for (int e = 0; e < 16; ++e)
      a.v[e] = (bf16_t)((float)fa.v[e] - (float)xa.v[e]);
#pragma unroll
    for (int n = 0; n < NB; ++n) {
      Frag bf;
      bf.q[0] = *(const uint4*)(wr[n] + k0);
      bf.q[1] = *(const uint4*)(wr[n] + k0 + 16);
      acc[n] = __builtin_amdgcn_wmma_f32_16x16x32_bf16(false, a.v, false, bf.v,
                                                       (short)0, acc[n], false, false);
    }
  }
#pragma unroll
  for (int n = 0; n < NB; ++n) {
    const float bv = bias[nbase + n * 16 + row];
    bf16_t* yb = Y + (size_t)(pbase + hi * 8) * Co + nbase + n * 16 + row;
    for (int r = 0; r < 8; ++r)
      yb[(size_t)r * Co] = (bf16_t)(acc[n][r] + bv);
  }
}

// ---------------------------------------------------------------------------
// Per-channel sum / sumsq (train-mode BN statistics). One block per channel.
// ---------------------------------------------------------------------------
__global__ void k_stats(const bf16_t* __restrict__ Y, float* __restrict__ stats,
                        int P, int C) {
  __shared__ float s1[256], s2[256];
  const int c = blockIdx.x;
  float a = 0.0f, b = 0.0f;
  for (int p = threadIdx.x; p < P; p += 256) {
    float v = (float)Y[(size_t)p * C + c];
    a += v; b += v * v;
  }
  s1[threadIdx.x] = a; s2[threadIdx.x] = b;
  __syncthreads();
  for (int off = 128; off > 0; off >>= 1) {
    if (threadIdx.x < off) {
      s1[threadIdx.x] += s1[threadIdx.x + off];
      s2[threadIdx.x] += s2[threadIdx.x + off];
    }
    __syncthreads();
  }
  if (threadIdx.x == 0) { stats[2*c] = s1[0]; stats[2*c+1] = s2[0]; }
}

// BN (train mode) + ReLU, in place on bf16.
__global__ void k_bn_relu(bf16_t* __restrict__ Y, const float* __restrict__ stats,
                          const float* __restrict__ g, const float* __restrict__ be,
                          size_t total, int C, float invP) {
  size_t i = (size_t)blockIdx.x * 256 + threadIdx.x;
  size_t stride = (size_t)gridDim.x * 256;
  for (; i < total; i += stride) {
    int c = (int)(i % C);
    float m   = stats[2*c] * invP;
    float var = stats[2*c+1] * invP - m * m;
    float sc  = g[c] * rsqrtf(var + 1e-5f);
    float sh  = be[c] - m * sc;
    float v = (float)Y[i] * sc + sh;
    Y[i] = (bf16_t)(v > 0.0f ? v : 0.0f);
  }
}

// ---------------------------------------------------------------------------
// Attention combine: y = sum_k( C * softmax_c(scores) * alpha ) / K.
// One block per (b,s); 256 threads = 256 channels.
// ---------------------------------------------------------------------------
__global__ void k_attn_combine(const bf16_t* __restrict__ scores,
                               const bf16_t* __restrict__ alpha,
                               float* __restrict__ out5) {
  __shared__ float sh[256];
  const int bs = blockIdx.x;
  const int c  = threadIdx.x;
  const size_t base = (size_t)bs * KK * 256;
  float acc = 0.0f;
  for (int k = 0; k < KK; ++k) {
    float sc = (float)scores[base + (size_t)k * 256 + c];
    sh[c] = sc; __syncthreads();
    for (int off = 128; off > 0; off >>= 1) {
      if (c < off) sh[c] = fmaxf(sh[c], sh[c + off]);
      __syncthreads();
    }
    float m = sh[0]; __syncthreads();
    float e = expf(sc - m);
    sh[c] = e; __syncthreads();
    for (int off = 128; off > 0; off >>= 1) {
      if (c < off) sh[c] += sh[c + off];
      __syncthreads();
    }
    float Z = sh[0]; __syncthreads();
    acc += 256.0f * (e / Z) * (float)alpha[base + (size_t)k * 256 + c];
  }
  out5[(size_t)bs * 320 + 64 + c] = acc * (1.0f / (float)KK);
}

// ---------------------------------------------------------------------------
// Host side
// ---------------------------------------------------------------------------
static inline void launch_gemm(const bf16_t* X, const bf16_t* W, const float* bias,
                               bf16_t* Y, int P, int Ci, int Co, hipStream_t s) {
  const int ntiles = Co >> 4;
  if ((ntiles & 3) == 0) {
    int waves = (P >> 4) * (ntiles / 4);
    k_gemm<4><<<waves / 8, 256, 0, s>>>(X, W, bias, Y, P, Ci, Co);
  } else {
    int waves = (P >> 4) * (ntiles / 2);
    k_gemm<2><<<waves / 8, 256, 0, s>>>(X, W, bias, Y, P, Ci, Co);
  }
}

static inline void launch_bn(bf16_t* Y, float* stats, const float* g, const float* be,
                             int P, int Co, hipStream_t s) {
  k_stats<<<Co, 256, 0, s>>>(Y, stats, P, Co);
  size_t total = (size_t)P * Co;
  int blocks = (int)((total + 255) / 256);
  k_bn_relu<<<blocks, 256, 0, s>>>(Y, stats, g, be, total, Co, 1.0f / (float)P);
}

extern "C" void kernel_launch(void* const* d_in, const int* in_sizes, int n_in,
                              void* d_out, int out_size, void* d_ws, size_t ws_size,
                              hipStream_t stream) {
  (void)in_sizes; (void)n_in; (void)out_size; (void)ws_size;
  const float* xyz   = (const float*)d_in[0];
  const float* eula  = (const float*)d_in[1];
  const float* nearb = (const float*)d_in[2];
  const float* meta  = (const float*)d_in[3];
  const float* fea   = (const float*)d_in[4];
  // mlp_params: [5]=W1,[6]=b1,[7]=g1,[8]=be1,[9]=W2,[10]=b2,[11]=g2,[12]=be2
  // attn heads fai/psi/alpha/gamma at 13+6h: W1,b1,g1,be1,W2,b2

  char* wsb = (char*)d_ws;
  size_t cur = 0;
  auto wsalloc = [&](size_t bytes) -> void* {
    cur = (cur + 255) & ~(size_t)255;
    void* p = wsb + cur;
    cur += bytes;
    return p;
  };

  int*    fps_idx  = (int*)   wsalloc((size_t)PC * 4);
  int*    knn      = (int*)   wsalloc((size_t)PC * KK * 4);
  bf16_t* X0       = (bf16_t*)wsalloc((size_t)PP * 96 * 2);
  bf16_t* Xc0      = (bf16_t*)wsalloc((size_t)PC * 96 * 2);
  bf16_t* H1       = (bf16_t*)wsalloc((size_t)PP * 128 * 2);
  bf16_t* Hc1      = (bf16_t*)wsalloc((size_t)PC * 128 * 2);
  bf16_t* H2       = (bf16_t*)wsalloc((size_t)PP * 256 * 2);   // new_fea
  bf16_t* Hc2      = (bf16_t*)wsalloc((size_t)PC * 256 * 2);   // cfa
  bf16_t* Hh       = (bf16_t*)wsalloc((size_t)PP * CH * 2);    // shared hidden
  bf16_t* Fh       = (bf16_t*)wsalloc((size_t)PC * CH * 2);
  bf16_t* fai_out  = (bf16_t*)wsalloc((size_t)PC * 256 * 2);
  bf16_t* psi_out  = (bf16_t*)wsalloc((size_t)PP * 256 * 2);   // later reused as scores
  bf16_t* alpha_out= (bf16_t*)wsalloc((size_t)PP * 256 * 2);
  float*  stats    = (float*) wsalloc(CH * 2 * 4);
  bf16_t* W1b      = (bf16_t*)wsalloc((size_t)128 * 96 * 2);
  bf16_t* W2b      = (bf16_t*)wsalloc((size_t)256 * 128 * 2);
  bf16_t* Wa1b[4]; bf16_t* Wa2b[4];
  float *a_b1[4], *a_g1[4], *a_be1[4];
  for (int h = 0; h < 4; ++h) {
    Wa1b[h] = (bf16_t*)wsalloc((size_t)CH * 256 * 2);
    Wa2b[h] = (bf16_t*)wsalloc((size_t)256 * CH * 2);
    a_b1[h]  = (float*)wsalloc(CH * 4);
    a_g1[h]  = (float*)wsalloc(CH * 4);
    a_be1[h] = (float*)wsalloc(CH * 4);
  }

  // ---- weight prep ----
  k_pack_w<<<(128*96 + 255)/256, 256, 0, stream>>>(W1b, (const float*)d_in[5], 128, 82, 128, 96);
  k_pack_w<<<(256*128 + 255)/256, 256, 0, stream>>>(W2b, (const float*)d_in[9], 256, 128, 256, 128);
  for (int h = 0; h < 4; ++h) {
    int base = 13 + 6 * h;
    k_pack_w<<<(CH*256 + 255)/256, 256, 0, stream>>>(Wa1b[h], (const float*)d_in[base],     264, 256, CH, 256);
    k_pack_w<<<(256*CH + 255)/256, 256, 0, stream>>>(Wa2b[h], (const float*)d_in[base + 4], 256, 264, 256, CH);
    k_pad_vec<<<(CH + 255)/256, 256, 0, stream>>>(a_b1[h],  (const float*)d_in[base + 1], 264, CH);
    k_pad_vec<<<(CH + 255)/256, 256, 0, stream>>>(a_g1[h],  (const float*)d_in[base + 2], 264, CH);
    k_pad_vec<<<(CH + 255)/256, 256, 0, stream>>>(a_be1[h], (const float*)d_in[base + 3], 264, CH);
  }

  // ---- sampling / grouping ----
  k_fps<<<BB, 256, 0, stream>>>(xyz, fps_idx);
  k_knn<<<PC / 256, 256, 0, stream>>>(xyz, fps_idx, knn);
  k_build_x0<<<PP / 256, 256, 0, stream>>>(xyz, eula, nearb, meta, fea, fps_idx, knn, X0);
  k_build_xc0<<<PC / 256, 256, 0, stream>>>(nearb, meta, fea, fps_idx, Xc0);
  k_outputs<<<PC / 256, 256, 0, stream>>>(xyz, eula, nearb, meta, fea, fps_idx, (float*)d_out);

  // ---- MLP layer 1 (82->128) + BN + ReLU, grouped and center paths ----
  launch_gemm(X0,  W1b, (const float*)d_in[6], H1,  PP, 96, 128, stream);
  launch_bn(H1, stats, (const float*)d_in[7], (const float*)d_in[8], PP, 128, stream);
  launch_gemm(Xc0, W1b, (const float*)d_in[6], Hc1, PC, 96, 128, stream);
  launch_bn(Hc1, stats, (const float*)d_in[7], (const float*)d_in[8], PC, 128, stream);

  // ---- MLP layer 2 (128->256) + BN + ReLU ----
  launch_gemm(H1,  W2b, (const float*)d_in[10], H2,  PP, 128, 256, stream);
  launch_bn(H2, stats, (const float*)d_in[11], (const float*)d_in[12], PP, 256, stream);
  launch_gemm(Hc1, W2b, (const float*)d_in[10], Hc2, PC, 128, 256, stream);
  launch_bn(Hc2, stats, (const float*)d_in[11], (const float*)d_in[12], PC, 256, stream);

  // ---- fai = mlp2(cfa) ----
  launch_gemm(Hc2, Wa1b[0], a_b1[0], Fh, PC, 256, CH, stream);
  launch_bn(Fh, stats, a_g1[0], a_be1[0], PC, CH, stream);
  launch_gemm(Fh, Wa2b[0], (const float*)d_in[18], fai_out, PC, CH, 256, stream);

  // ---- psi = mlp2(new_fea) ----
  launch_gemm(H2, Wa1b[1], a_b1[1], Hh, PP, 256, CH, stream);
  launch_bn(Hh, stats, a_g1[1], a_be1[1], PP, CH, stream);
  launch_gemm(Hh, Wa2b[1], (const float*)d_in[24], psi_out, PP, CH, 256, stream);

  // ---- alpha = mlp2(new_fea) ----
  launch_gemm(H2, Wa1b[2], a_b1[2], Hh, PP, 256, CH, stream);
  launch_bn(Hh, stats, a_g1[2], a_be1[2], PP, CH, stream);
  launch_gemm(Hh, Wa2b[2], (const float*)d_in[30], alpha_out, PP, CH, 256, stream);

  // ---- gamma = mlp2(fai - psi): fused subtract in A-fragment load ----
  {
    int waves = (PP >> 4) * ((CH >> 4) / 2);
    k_gemm_sub<2><<<waves / 8, 256, 0, stream>>>(psi_out, fai_out, Wa1b[3], a_b1[3],
                                                 Hh, PP, 256, CH);
  }
  launch_bn(Hh, stats, a_g1[3], a_be1[3], PP, CH, stream);
  launch_gemm(Hh, Wa2b[3], (const float*)d_in[36], psi_out /*scores*/, PP, CH, 256, stream);

  // ---- softmax over channels, weight by alpha, mean over K ----
  k_attn_combine<<<PC, 256, 0, stream>>>(psi_out, alpha_out, (float*)d_out + 98304);
}
